// FullereneNet_50190987821084
// MI455X (gfx1250) — compile-verified, matrix operand order checked by reading, not run on hardware
//
#include <hip/hip_runtime.h>
#include <math.h>

typedef __attribute__((ext_vector_type(16))) _Float16 v16h;
typedef __attribute__((ext_vector_type(8)))  float    v8f;

#define WMMA_F16(acc, a, b) \
  (acc) = __builtin_amdgcn_wmma_f32_16x16x32_f16(false, (a), false, (b), (short)0, (acc), false, false)

#define NN   10000
#define EE_  160000
#define GG   160
#define FIN  92
#define FE_  41
#define DD   64
#define HH   4
#define CC   64
#define HC_  256
#define LL   5
#define HID_ 128

// ---------------- WMMA fragment helpers (CDNA5 16-bit A/B layouts, ISA 7.12.2) ----
__device__ __forceinline__ int kmap16(int e, int half) {
  int i = e >> 1, j = e & 1;
  return (i < 4) ? (half * 8 + 2 * i + j) : (16 + half * 8 + 2 * (i - 4) + j);
}

// A fragment: 16x32 f16 tile at column kbase of a row-major f16 LDS array.
// Per lane this is two contiguous 8xf16 runs -> 2x ds_load_b128.
__device__ __forceinline__ v16h afrag_lds(const _Float16* A, int lda, int kbase) {
  int lane = threadIdx.x & 31;
  int m = lane & 15, half = lane >> 4;
  v16h a;
#pragma unroll
  for (int e = 0; e < 16; ++e) a[e] = A[m * lda + kbase + kmap16(e, half)];
  return a;
}

// B fragment from pre-packed fragment-order buffer: one 32B contiguous load per lane.
__device__ __forceinline__ v16h bfrag_p(const _Float16* P, int kt, int nt, int ntiles) {
  const _Float16* p = P + ((size_t)(kt * ntiles + nt)) * 512 + (threadIdx.x & 31) * 16;
  return *(const v16h*)p;
}

__device__ __forceinline__ float siluf(float x) { return x / (1.0f + __expf(-x)); }
__device__ __forceinline__ float softplusf(float x) {
  return (x > 20.0f) ? x : log1pf(__expf(x));
}
__device__ __forceinline__ void atomicMaxF(float* addr, float val) {
  unsigned int* ua = reinterpret_cast<unsigned int*>(addr);
  unsigned int cur = *ua;
  while (__uint_as_float(cur) < val) {
    unsigned int prev = atomicCAS(ua, cur, __float_as_uint(val));
    if (prev == cur) break;
    cur = prev;
  }
}

// ---------------- weight packing: f32 [K,N] row-major -> f16 WMMA-B fragment order --
// Tile (kt,nt) holds 32 lanes x 16 f16, lane-contiguous; zero-pad rows k >= K.
__global__ void pack_b_kernel(const float* __restrict__ B, _Float16* __restrict__ P,
                              int K, int N) {
  int ktiles = (K + 31) >> 5, ntiles = N >> 4;
  int tid = blockIdx.x * blockDim.x + threadIdx.x;
  if (tid >= ktiles * ntiles * 32) return;
  int lane = tid & 31, tile = tid >> 5;
  int kt = tile / ntiles, nt = tile - kt * ntiles;
  int n = nt * 16 + (lane & 15), half = lane >> 4;
  _Float16* out = P + (size_t)tile * 512 + lane * 16;
#pragma unroll
  for (int e = 0; e < 16; ++e) {
    int k = kt * 32 + kmap16(e, half);
    int kc = (k < K) ? k : K - 1;
    float v = B[(long)kc * N + n];
    out[e] = (k < K) ? (_Float16)v : (_Float16)0.0f;
  }
}

// ---------------- generic GEMM: C[M,Ncol] = act(A[M,K] @ B[K,Ncol] + bias) ----------
// block = 128 threads (4 waves); 16(M) x 64(N) tile per block; B pre-packed f16.
__global__ void wmma_gemm_kernel(const float* __restrict__ A, const _Float16* __restrict__ Bp,
                                 const float* __restrict__ bias, float* __restrict__ C,
                                 int M, int K, int Ncol, int act) {
  __shared__ __align__(32) _Float16 sA[16 * 256];  // K <= 256 in this model
  int mbase = blockIdx.x * 16;
  int wave = threadIdx.x >> 5, lane = threadIdx.x & 31;
  int Kpad = (K + 31) & ~31;
  int ntiles = Ncol >> 4;
  // branchless zero-padded staging (unconditional loads + cndmask)
  for (int idx = threadIdx.x; idx < 16 * Kpad; idx += blockDim.x) {
    int r = idx / Kpad, k = idx - r * Kpad;
    int gr = mbase + r;
    int grc = (gr < M) ? gr : M - 1;
    int kcc = (k < K) ? k : K - 1;
    float v = A[(long)grc * K + kcc];
    sA[r * Kpad + k] = (gr < M && k < K) ? (_Float16)v : (_Float16)0.0f;
  }
  __syncthreads();
  int nt = (blockIdx.y << 2) + wave;          // this wave's 16-col tile
  v8f acc = {};
  for (int kt = 0; kt < (Kpad >> 5); ++kt) {
    v16h a = afrag_lds(sA, Kpad, kt * 32);
    v16h b = bfrag_p(Bp, kt, nt, ntiles);
    WMMA_F16(acc, a, b);
  }
  int n = lane & 15, half = lane >> 4;
  int col = nt * 16 + n;
  float bv = bias ? bias[col] : 0.0f;
#pragma unroll
  for (int r = 0; r < 8; ++r) {
    int row = mbase + half * 8 + r;
    if (row < M) {
      float v = acc[r] + bv;
      if (act == 1) v = siluf(v);
      else if (act == 2) v = softplusf(v);
      C[(long)row * Ncol + col] = v;
    }
  }
}

// ---------------- fused per-edge attention score kernel -----------------------------
// 128 threads, 16 edges per block. ee=e@We+be, k+=ee, pre1/pre2 [64,64]@[64,64] WMMA,
// tanh/softplus reductions, q.k term, alpha_raw + per-(node,head) running max.
__global__ void edge_attn_kernel(const float* __restrict__ Q, const float* __restrict__ Kn,
                                 const float* __restrict__ eF,
                                 const _Float16* __restrict__ WeP, const float* __restrict__ be,
                                 const _Float16* __restrict__ A1WP, const float* __restrict__ A1b,
                                 const float* __restrict__ A1v,
                                 const _Float16* __restrict__ A2WP, const float* __restrict__ A2b,
                                 const float* __restrict__ A2v,
                                 const int* __restrict__ src, const int* __restrict__ dst,
                                 float* __restrict__ alphaRaw, float* __restrict__ nodeMax,
                                 int Etot) {
  __shared__ __align__(32) _Float16 sQ[16 * 256];
  __shared__ __align__(32) _Float16 sK[16 * 256];
  __shared__ __align__(32) _Float16 sE[16 * 256];
  __shared__ __align__(32) _Float16 sF[16 * 64];
  __shared__ float sA1[64], sA2[64];
  __shared__ int sSrc[16], sDst[16];
  int t = threadIdx.x, wave = t >> 5, lane = t & 31;
  int e0 = blockIdx.x * 16;
  if (t < 16) {
    int e = (e0 + t < Etot) ? e0 + t : Etot - 1;
    sSrc[t] = src[e]; sDst[t] = dst[e];
  }
  if (t < 64) { sA1[t] = 0.0f; sA2[t] = 0.0f; }
  __syncthreads();
  for (int idx = t; idx < 16 * 256; idx += 128) {
    int r = idx >> 8, c = idx & 255;
    sQ[idx] = (_Float16)Q[(long)sDst[r] * 256 + c];
    sK[idx] = (_Float16)Kn[(long)sSrc[r] * 256 + c];
  }
  for (int idx = t; idx < 16 * 64; idx += 128) {
    int r = idx >> 6, c = idx & 63;
    int e = (e0 + r < Etot) ? e0 + r : Etot - 1;
    sF[idx] = (_Float16)eF[(long)e * 64 + c];
  }
  __syncthreads();
  // ee = sF @ We + be; wave w handles column tiles 4w..4w+3 (We packed: ntiles=16)
  for (int i = 0; i < 4; ++i) {
    int nt = wave * 4 + i;
    v8f acc = {};
#pragma unroll
    for (int kt = 0; kt < 2; ++kt) {
      v16h a = afrag_lds(sF, 64, kt * 32);
      v16h b = bfrag_p(WeP, kt, nt, 16);
      WMMA_F16(acc, a, b);
    }
    int n = lane & 15, half = lane >> 4, col = nt * 16 + n;
    float bv = be[col];
#pragma unroll
    for (int r = 0; r < 8; ++r) {
      int row = half * 8 + r;
      float v = acc[r] + bv;
      sE[row * 256 + col] = (_Float16)v;
      sK[row * 256 + col] = (_Float16)((float)sK[row * 256 + col] + v);  // k += ee
    }
  }
  __syncthreads();
  // pre1/pre2 over [64 rows=(edge,head), 64 cols]; wave w owns row-tile w.
  // A1W/A2W packed as [192,64] (ntiles=4, ktiles=6): q rows kt+0, k rows kt+2, e rows kt+4.
  int mt = wave;
  for (int nt = 0; nt < 4; ++nt) {
    v8f a1acc = {}, a2acc = {};
#pragma unroll
    for (int kt = 0; kt < 2; ++kt) {
      v16h aq = afrag_lds(sQ + mt * 1024, 64, kt * 32);
      v16h ak = afrag_lds(sK + mt * 1024, 64, kt * 32);
      v16h ae = afrag_lds(sE + mt * 1024, 64, kt * 32);
      WMMA_F16(a1acc, aq, bfrag_p(A1WP, kt, nt, 4));
      WMMA_F16(a1acc, ak, bfrag_p(A1WP, kt + 2, nt, 4));
      WMMA_F16(a1acc, ae, bfrag_p(A1WP, kt + 4, nt, 4));
      WMMA_F16(a2acc, aq, bfrag_p(A2WP, kt, nt, 4));
      WMMA_F16(a2acc, ak, bfrag_p(A2WP, kt + 2, nt, 4));
      WMMA_F16(a2acc, ae, bfrag_p(A2WP, kt + 4, nt, 4));
    }
    int n = lane & 15, half = lane >> 4, col = nt * 16 + n;
    float b1vv = A1b[col], b2vv = A2b[col], w1 = A1v[col], w2 = A2v[col];
#pragma unroll
    for (int r = 0; r < 8; ++r) {
      int row = mt * 16 + half * 8 + r;
      atomicAdd(&sA1[row], tanhf(a1acc[r] + b1vv) * w1);
      atomicAdd(&sA2[row], softplusf(a2acc[r] + b2vv) * w2);
    }
  }
  __syncthreads();
  if (t < 64) {
    const _Float16* qr = sQ + t * 64;
    const _Float16* kr = sK + t * 64;
    float s = 0.0f;
    for (int c = 0; c < 64; ++c) s += (float)qr[c] * (float)kr[c];
    float x = sA1[t] + sA2[t] + s * 0.125f;   // mult = q.k / sqrt(64)
    float al = siluf(x);
    int edge = t >> 2, head = t & 3;
    int ge = e0 + edge;
    if (ge < Etot) {
      alphaRaw[(long)ge * 4 + head] = al;
      atomicMaxF(&nodeMax[sDst[edge] * 4 + head], al);
    }
  }
}

// ---------------- segment-softmax: exp(a - max) and denominator ---------------------
__global__ void softmax_edge_kernel(float* __restrict__ alphaRaw, const float* __restrict__ nodeMax,
                                    float* __restrict__ denom, const int* __restrict__ dst, int Etot) {
  int i = blockIdx.x * blockDim.x + threadIdx.x;
  if (i >= Etot * 4) return;
  int e = i >> 2, h = i & 3;
  float v = __expf(alphaRaw[i] - nodeMax[dst[e] * 4 + h]);
  alphaRaw[i] = v;
  atomicAdd(&denom[dst[e] * 4 + h], v);
}

// ---------------- fused aggregate: m=(v+ee)*alpha, per-edge LN, scatter-add ---------
__global__ void edge_aggregate_kernel(const float* __restrict__ V, const float* __restrict__ eF,
                                      const _Float16* __restrict__ WeP, const float* __restrict__ be,
                                      const float* __restrict__ alphaE, const float* __restrict__ denom,
                                      const float* __restrict__ lng, const float* __restrict__ lnb,
                                      const int* __restrict__ src, const int* __restrict__ dst,
                                      float* __restrict__ agg, int Etot) {
  __shared__ float sM[16 * 256];
  __shared__ __align__(32) _Float16 sF[16 * 64];
  __shared__ float sAN[64];
  __shared__ float sMu[16], sRs[16];
  __shared__ int sSrc[16], sDst[16];
  int t = threadIdx.x, wave = t >> 5, lane = t & 31;
  int e0 = blockIdx.x * 16;
  if (t < 16) {
    int e = (e0 + t < Etot) ? e0 + t : Etot - 1;
    sSrc[t] = src[e]; sDst[t] = dst[e];
  }
  __syncthreads();
  if (t < 64) {
    int ed = t >> 2, hh = t & 3;
    int e = (e0 + ed < Etot) ? e0 + ed : Etot - 1;
    sAN[t] = alphaE[(long)e * 4 + hh] / (denom[sDst[ed] * 4 + hh] + 1e-16f);
  }
  for (int idx = t; idx < 16 * 256; idx += 128) {
    int r = idx >> 8, c = idx & 255;
    sM[idx] = V[(long)sSrc[r] * 256 + c];
  }
  for (int idx = t; idx < 16 * 64; idx += 128) {
    int r = idx >> 6, c = idx & 63;
    int e = (e0 + r < Etot) ? e0 + r : Etot - 1;
    sF[idx] = (_Float16)eF[(long)e * 64 + c];
  }
  __syncthreads();
  for (int i = 0; i < 4; ++i) {
    int nt = wave * 4 + i;
    v8f acc = {};
#pragma unroll
    for (int kt = 0; kt < 2; ++kt) {
      v16h a = afrag_lds(sF, 64, kt * 32);
      v16h b = bfrag_p(WeP, kt, nt, 16);
      WMMA_F16(acc, a, b);
    }
    int n = lane & 15, half = lane >> 4, col = nt * 16 + n;
    float bv = be[col];
    int hh = col >> 6;
#pragma unroll
    for (int r = 0; r < 8; ++r) {
      int row = half * 8 + r;
      sM[row * 256 + col] = (sM[row * 256 + col] + acc[r] + bv) * sAN[row * 4 + hh];
    }
  }
  __syncthreads();
  if (t < 16) {
    float s = 0.0f, s2 = 0.0f;
    for (int c = 0; c < 256; ++c) { float v = sM[t * 256 + c]; s += v; s2 += v * v; }
    float mu = s * (1.0f / 256.0f);
    float var = s2 * (1.0f / 256.0f) - mu * mu;
    sMu[t] = mu; sRs[t] = rsqrtf(var + 1e-5f);
  }
  __syncthreads();
  for (int idx = t; idx < 16 * 256; idx += 128) {
    int r = idx >> 8, c = idx & 255;
    int ge = e0 + r;
    if (ge < Etot) {
      float v = (sM[idx] - sMu[r]) * sRs[r] * lng[c] + lnb[c];
      atomicAdd(&agg[(long)sDst[r] * 256 + c], v);
    }
  }
}

// ---------------- BN stats / apply, pooling, head ----------------------------------
__global__ void fill_kernel(float* p, float v, int n) {
  int i = blockIdx.x * blockDim.x + threadIdx.x;
  if (i < n) p[i] = v;
}

__global__ void colstats_kernel(const float* __restrict__ o, float* __restrict__ sum,
                                float* __restrict__ sq, int M) {
  int tid = blockIdx.x * blockDim.x + threadIdx.x;  // 64 blocks * 256
  int c = tid & 63, r0 = tid >> 6;
  float a = 0.0f, a2 = 0.0f;
  for (int r = r0; r < M; r += 256) {
    float v = o[(long)r * 64 + c];
    a += v; a2 += v * v;
  }
  atomicAdd(&sum[c], a);
  atomicAdd(&sq[c], a2);
}

__global__ void bn_apply_kernel(const float* __restrict__ o, const float* __restrict__ sum,
                                const float* __restrict__ sq, const float* __restrict__ g,
                                const float* __restrict__ b, const float* __restrict__ hw,
                                float* __restrict__ hout, int M) {
  int i = blockIdx.x * blockDim.x + threadIdx.x;
  if (i >= M * 64) return;
  int c = i & 63;
  float inv = 1.0f / (float)M;
  float mu = sum[c] * inv;
  float var = sq[c] * inv - mu * mu;
  float v = (o[i] - mu) * rsqrtf(var + 1e-5f) * g[c] + b[c];
  hout[i] = siluf(v) + hw[i];
}

__global__ void pool_accum_kernel(const float* __restrict__ h, const int* __restrict__ batch,
                                  float* __restrict__ psum, float* __restrict__ pcnt, int M) {
  int i = blockIdx.x * blockDim.x + threadIdx.x;
  if (i >= M * 64) return;
  int r = i >> 6, c = i & 63;
  int g = batch[r];
  atomicAdd(&psum[g * 64 + c], h[i]);
  if (c == 0) atomicAdd(&pcnt[g], 1.0f);
}

__global__ void pool_finalize_kernel(const float* __restrict__ psum, const float* __restrict__ pcnt,
                                     float* __restrict__ pooled, int Gn) {
  int i = blockIdx.x * blockDim.x + threadIdx.x;
  if (i >= Gn * 64) return;
  pooled[i] = psum[i] / fmaxf(pcnt[i >> 6], 1.0f);
}

__global__ void out_kernel(const float* __restrict__ hid, const float* __restrict__ outW,
                           const float* __restrict__ outb, float* __restrict__ out, int Gn) {
  int g = blockIdx.x * blockDim.x + threadIdx.x;
  if (g >= Gn) return;
  float s = outb[0];
  for (int c = 0; c < HID_; ++c) s += hid[g * HID_ + c] * outW[c];
  out[g] = s;
}

// ---------------- host orchestration -----------------------------------------------
static inline void run_gemm(const float* A, const _Float16* Bp, const float* bias, float* C,
                            int M, int K, int Ncol, int act, hipStream_t s) {
  dim3 grid((M + 15) / 16, Ncol / 64);
  wmma_gemm_kernel<<<grid, 128, 0, s>>>(A, Bp, bias, C, M, K, Ncol, act);
}
static inline void run_fill(float* p, float v, int n, hipStream_t s) {
  fill_kernel<<<(n + 255) / 256, 256, 0, s>>>(p, v, n);
}
static inline size_t packed_elems(int K, int N) {
  return (size_t)((K + 31) / 32) * (N / 16) * 512;
}
static inline void run_pack(const float* B, _Float16* P, int K, int N, hipStream_t s) {
  int threads = ((K + 31) / 32) * (N / 16) * 32;
  pack_b_kernel<<<(threads + 127) / 128, 128, 0, s>>>(B, P, K, N);
}

extern "C" void kernel_launch(void* const* d_in, const int* in_sizes, int n_in,
                              void* d_out, int out_size, void* d_ws, size_t ws_size,
                              hipStream_t stream) {
  (void)in_sizes; (void)n_in; (void)out_size; (void)ws_size;
  const float* x         = (const float*)d_in[0];
  const float* edge_attr = (const float*)d_in[1];
  const int*   eidx      = (const int*)d_in[2];
  const int*   batch     = (const int*)d_in[3];
  const float* emb_W = (const float*)d_in[4];   const float* emb_b = (const float*)d_in[5];
  const float* ee_W1 = (const float*)d_in[6];   const float* ee_b1 = (const float*)d_in[7];
  const float* ee_W2 = (const float*)d_in[8];   const float* ee_b2 = (const float*)d_in[9];
  const float* Wq = (const float*)d_in[10];  const float* bq = (const float*)d_in[11];
  const float* Wk = (const float*)d_in[12];  const float* bk = (const float*)d_in[13];
  const float* Wv = (const float*)d_in[14];  const float* bv = (const float*)d_in[15];
  const float* We = (const float*)d_in[16];  const float* be = (const float*)d_in[17];
  const float* A1W = (const float*)d_in[18]; const float* A1b = (const float*)d_in[19];
  const float* A1v = (const float*)d_in[20];
  const float* A2W = (const float*)d_in[21]; const float* A2b = (const float*)d_in[22];
  const float* A2v = (const float*)d_in[23];
  const float* Wc = (const float*)d_in[24];  const float* bc = (const float*)d_in[25];
  const float* bng = (const float*)d_in[26]; const float* bnb = (const float*)d_in[27];
  const float* lng = (const float*)d_in[28]; const float* lnb = (const float*)d_in[29];
  const float* Wf = (const float*)d_in[30];  const float* bf = (const float*)d_in[31];
  const float* fcW = (const float*)d_in[32]; const float* fcb = (const float*)d_in[33];
  const float* outW = (const float*)d_in[34]; const float* outb = (const float*)d_in[35];

  const int* src = eidx;
  const int* dst = eidx + EE_;

  // -------- float workspace carve-up --------
  float* ws = (float*)d_ws;
  size_t off = 0;
  float* h0   = ws + off; off += (size_t)NN * DD;
  float* h1   = ws + off; off += (size_t)NN * DD;
  float* eF   = ws + off; off += (size_t)EE_ * DD;
  float* Qn   = ws + off; off += (size_t)NN * HC_;
  float* Kn   = ws + off; off += (size_t)NN * HC_;
  float* Vn   = ws + off; off += (size_t)NN * HC_;
  float* agg  = ws + off; off += (size_t)NN * HC_;
  float* alphaRaw = ws + off; off += (size_t)EE_ * HH;
  float* nodeMax  = ws + off; off += (size_t)NN * HH;
  float* denom    = ws + off; off += (size_t)NN * HH;
  float* oB   = ws + off; off += (size_t)NN * DD;
  float* hwB  = ws + off; off += (size_t)NN * DD;
  float* bnSum = ws + off; off += DD;
  float* bnSq  = ws + off; off += DD;
  float* psum  = ws + off; off += (size_t)GG * DD;
  float* pcnt  = ws + off; off += GG;
  float* pooled = ws + off; off += (size_t)GG * DD;
  float* hid   = ws + off; off += (size_t)GG * HID_;
  float* tmpE = Qn;  // edge-embed temp overlays Qn..agg (E*64 == N*256*4), pre-layer only

  // -------- packed f16 weights (fragment order), after float region (32B aligned) ---
  _Float16* fp = (_Float16*)(ws + off);
  size_t foff = 0;
  _Float16* embP  = fp + foff; foff += packed_elems(FIN, DD);
  _Float16* eW1P  = fp + foff; foff += packed_elems(FE_, DD);
  _Float16* eW2P  = fp + foff; foff += packed_elems(DD, DD);
  _Float16* fcP   = fp + foff; foff += packed_elems(DD, HID_);
  _Float16 *WqP[LL], *WkP[LL], *WvP[LL], *WeP[LL], *A1WP[LL], *A2WP[LL], *WcP[LL], *WfP[LL];
  for (int l = 0; l < LL; ++l) {
    WqP[l]  = fp + foff; foff += packed_elems(DD, HC_);
    WkP[l]  = fp + foff; foff += packed_elems(DD, HC_);
    WvP[l]  = fp + foff; foff += packed_elems(DD, HC_);
    WeP[l]  = fp + foff; foff += packed_elems(DD, HC_);
    A1WP[l] = fp + foff; foff += packed_elems(3 * CC, CC);
    A2WP[l] = fp + foff; foff += packed_elems(3 * CC, CC);
    WcP[l]  = fp + foff; foff += packed_elems(HC_, DD);
    WfP[l]  = fp + foff; foff += packed_elems(DD, DD);
  }

  // pack all weights
  run_pack(emb_W, embP, FIN, DD, stream);
  run_pack(ee_W1, eW1P, FE_, DD, stream);
  run_pack(ee_W2, eW2P, DD, DD, stream);
  run_pack(fcW, fcP, DD, HID_, stream);
  for (int l = 0; l < LL; ++l) {
    run_pack(Wq + (size_t)l * DD * HC_, WqP[l], DD, HC_, stream);
    run_pack(Wk + (size_t)l * DD * HC_, WkP[l], DD, HC_, stream);
    run_pack(Wv + (size_t)l * DD * HC_, WvP[l], DD, HC_, stream);
    run_pack(We + (size_t)l * DD * HC_, WeP[l], DD, HC_, stream);
    run_pack(A1W + (size_t)l * 3 * CC * CC, A1WP[l], 3 * CC, CC, stream);
    run_pack(A2W + (size_t)l * 3 * CC * CC, A2WP[l], 3 * CC, CC, stream);
    run_pack(Wc + (size_t)l * HC_ * DD, WcP[l], HC_, DD, stream);
    run_pack(Wf + (size_t)l * DD * DD, WfP[l], DD, DD, stream);
  }

  // embedding + edge embedding
  run_gemm(x, embP, emb_b, h0, NN, FIN, DD, 0, stream);
  run_gemm(edge_attr, eW1P, ee_b1, tmpE, EE_, FE_, DD, 2, stream);   // softplus
  run_gemm(tmpE, eW2P, ee_b2, eF, EE_, DD, DD, 0, stream);

  float* hc = h0;
  float* hn = h1;
  for (int l = 0; l < LL; ++l) {
    const float* bq_l = bq + (size_t)l * HC_;
    const float* bk_l = bk + (size_t)l * HC_;
    const float* bv_l = bv + (size_t)l * HC_;
    const float* be_l = be + (size_t)l * HC_;
    const float* A1b_l = A1b + (size_t)l * CC;  const float* A1v_l = A1v + (size_t)l * CC;
    const float* A2b_l = A2b + (size_t)l * CC;  const float* A2v_l = A2v + (size_t)l * CC;
    const float* bc_l = bc + (size_t)l * DD;
    const float* bng_l = bng + (size_t)l * DD;  const float* bnb_l = bnb + (size_t)l * DD;
    const float* lng_l = lng + (size_t)l * HC_; const float* lnb_l = lnb + (size_t)l * HC_;
    const float* bf_l = bf + (size_t)l * DD;

    run_gemm(hc, WqP[l], bq_l, Qn, NN, DD, HC_, 0, stream);
    run_gemm(hc, WkP[l], bk_l, Kn, NN, DD, HC_, 0, stream);
    run_gemm(hc, WvP[l], bv_l, Vn, NN, DD, HC_, 0, stream);

    run_fill(nodeMax, -3.0e38f, NN * HH, stream);
    run_fill(denom, 0.0f, NN * HH, stream);
    run_fill(agg, 0.0f, NN * HC_, stream);

    edge_attn_kernel<<<EE_ / 16, 128, 0, stream>>>(
        Qn, Kn, eF, WeP[l], be_l, A1WP[l], A1b_l, A1v_l, A2WP[l], A2b_l, A2v_l,
        src, dst, alphaRaw, nodeMax, EE_);
    softmax_edge_kernel<<<(EE_ * HH + 255) / 256, 256, 0, stream>>>(
        alphaRaw, nodeMax, denom, dst, EE_);
    edge_aggregate_kernel<<<EE_ / 16, 128, 0, stream>>>(
        Vn, eF, WeP[l], be_l, alphaRaw, denom, lng_l, lnb_l, src, dst, agg, EE_);

    run_gemm(agg, WcP[l], bc_l, oB, NN, HC_, DD, 0, stream);
    run_gemm(hc, WfP[l], bf_l, hwB, NN, DD, DD, 0, stream);

    run_fill(bnSum, 0.0f, DD, stream);
    run_fill(bnSq, 0.0f, DD, stream);
    colstats_kernel<<<64, 256, 0, stream>>>(oB, bnSum, bnSq, NN);
    bn_apply_kernel<<<(NN * DD + 255) / 256, 256, 0, stream>>>(
        oB, bnSum, bnSq, bng_l, bnb_l, hwB, hn, NN);

    float* t = hc; hc = hn; hn = t;
  }

  // global mean pool + head
  run_fill(psum, 0.0f, GG * DD, stream);
  run_fill(pcnt, 0.0f, GG, stream);
  pool_accum_kernel<<<(NN * DD + 255) / 256, 256, 0, stream>>>(hc, batch, psum, pcnt, NN);
  pool_finalize_kernel<<<(GG * DD + 255) / 256, 256, 0, stream>>>(psum, pcnt, pooled, GG);
  run_gemm(pooled, fcP, fcb, hid, GG, DD, HID_, 1, stream);  // silu
  out_kernel<<<(GG + 255) / 256, 256, 0, stream>>>(hid, outW, outb, (float*)d_out, GG);
}
